// MultiScaleSparseTransformerBlockV2_76166950028346
// MI455X (gfx1250) — matrix operand order repaired and checked by reading, not compiled, not run on hardware
//
#include <hip/hip_runtime.h>
#include <stdint.h>

// ---------------------------------------------------------------------------
// MI455X (gfx1250) implementation.
// - All GEMMs: v_wmma_f32_16x16x32_bf16, 4 output tiles per wave (A- or
//   B-fragment reuse across sub-tiles -> 4 WMMAs per K-step).
// - Attention: flash-style, block-sparse skip via ballot, K/V tiles staged
//   to LDS with GLOBAL_LOAD_ASYNC_TO_LDS_B128 (ASYNCcnt) when available.
// ---------------------------------------------------------------------------

typedef __attribute__((ext_vector_type(16))) __bf16 v16bf;
typedef __attribute__((ext_vector_type(8)))  float  v8f;
typedef __attribute__((ext_vector_type(4)))  int    v4i;
typedef __attribute__((address_space(1))) v4i* gv4i_p;   // global AS pointer
typedef __attribute__((address_space(3))) v4i* lv4i_p;   // LDS AS pointer

#define BATCH 4
#define CH    256
#define HH    48
#define WW    48
#define HWSZ  (HH*WW)          // 2304
#define SEQ   2484             // 2304 + 144 + 36
#define SPAD  2496             // padded to multiple of 32
#define LOCAL 2304
#define NHEAD 8
#define DHEAD 32
#define BOT   64

#if defined(__has_builtin)
# if __has_builtin(__builtin_amdgcn_global_load_async_to_lds_b128) && \
     __has_builtin(__builtin_amdgcn_s_wait_asynccnt)
#  define HAVE_ASYNC_LDS 1
# endif
#endif
#ifndef HAVE_ASYNC_LDS
# define HAVE_ASYNC_LDS 0
#endif

// ---------------- wave helpers ----------------
__device__ inline float wave_sum32(float v) {
  #pragma unroll
  for (int o = 16; o > 0; o >>= 1) v += __shfl_xor(v, o);
  return v;
}

// ---- 16 x (NT*16) tiles:  D = A(bf16, MxK) * W(f32, NxK)^T  ---------------
// A-fragment loaded once per K-step and reused across NT WMMAs.
template <int NT>
__device__ inline void gemm_tiles_bf16_wf32(const __bf16* __restrict__ A, int lda,
                                            const float*  __restrict__ W, int ldw,
                                            int m0, int n0, int K, v8f* acc) {
  const int lane = threadIdx.x & 31;
  const int half = lane >> 4;     // lane group g
  const int idx  = lane & 15;
  const __bf16* arow = A + (size_t)(m0 + idx) * lda;
  const float* wrow[NT];
  #pragma unroll
  for (int t = 0; t < NT; ++t) wrow[t] = W + (size_t)(n0 + t*16 + idx) * ldw;
  for (int k0 = 0; k0 < K; k0 += 32) {
    v16bf a;
    #pragma unroll
    for (int i = 0; i < 8; ++i) {            // A frag: k = g*8+i / 16+g*8+i
      a[i]     = arow[k0 + half*8 + i];
      a[i + 8] = arow[k0 + 16 + half*8 + i];
    }
    #pragma unroll
    for (int t = 0; t < NT; ++t) {
      v16bf b;
      #pragma unroll
      for (int i = 0; i < 16; ++i)           // B frag: k = g*8+i / 16+g*8+i, col = idx
        b[i] = (__bf16)wrow[t][k0 + (i >> 3)*16 + half*8 + (i & 7)];
      acc[t] = __builtin_amdgcn_wmma_f32_16x16x32_bf16(false, a, false, b,
                                                       (short)0, acc[t], false, false);
    }
  }
}

// ---------------- 1x1 conv:  xp[b][o][s] = sum_c W[o][c]*x[b][c][s] + bias -
// 64x16 output per wave: the strided x-column B-fragment is loaded once per
// K-step and reused across 4 M sub-tiles of conv_w.
__global__ __launch_bounds__(32)
void conv1x1_kernel(const float* __restrict__ x, const float* __restrict__ w,
                    const float* __restrict__ bias, float* __restrict__ xp) {
  const int lane = threadIdx.x, half = lane >> 4, idx = lane & 15;
  const int n0 = blockIdx.x * 16, m0 = blockIdx.y * 64, b = blockIdx.z;
  const float* xb = x + (size_t)b * CH * HWSZ;
  v8f acc[4] = {};
  for (int k0 = 0; k0 < CH; k0 += 32) {
    v16bf bb;
    #pragma unroll
    for (int i = 0; i < 16; ++i)
      bb[i] = (__bf16)xb[(size_t)(k0 + (i >> 3)*16 + half*8 + (i & 7)) * HWSZ + n0 + idx];
    #pragma unroll
    for (int mt = 0; mt < 4; ++mt) {
      v16bf a;
      const float* arow = w + (size_t)(m0 + mt*16 + idx) * CH + k0;
      #pragma unroll
      for (int i = 0; i < 8; ++i) {
        a[i]     = (__bf16)arow[half*8 + i];
        a[i + 8] = (__bf16)arow[16 + half*8 + i];
      }
      acc[mt] = __builtin_amdgcn_wmma_f32_16x16x32_bf16(false, a, false, bb,
                                                        (short)0, acc[mt], false, false);
    }
  }
  float* xpb = xp + (size_t)b * CH * HWSZ;
  #pragma unroll
  for (int mt = 0; mt < 4; ++mt)
    #pragma unroll
    for (int r = 0; r < 8; ++r) {
      const int o = m0 + mt*16 + half*8 + r;
      xpb[(size_t)o * HWSZ + n0 + idx] = acc[mt][r] + bias[o];
    }
}

// ---------------- LayerNorm over C of one xp column -> f (bf16) -----------
__global__ __launch_bounds__(32)
void ln_local_kernel(const float* __restrict__ xp, const float* __restrict__ w,
                     const float* __restrict__ bb, __bf16* __restrict__ f) {
  const int lane = threadIdx.x;
  const int s = blockIdx.x, b = blockIdx.y;
  const float* col = xp + (size_t)b * CH * HWSZ + s;
  float v[8], sum = 0.f;
  #pragma unroll
  for (int j = 0; j < 8; ++j) { v[j] = col[(size_t)(lane + 32*j) * HWSZ]; sum += v[j]; }
  const float mean = wave_sum32(sum) * (1.f/CH);
  float vs = 0.f;
  #pragma unroll
  for (int j = 0; j < 8; ++j) { float d = v[j] - mean; vs += d*d; }
  const float rstd = rsqrtf(wave_sum32(vs) * (1.f/CH) + 1e-5f);
  __bf16* fr = f + ((size_t)b * SPAD + s) * CH;
  #pragma unroll
  for (int j = 0; j < 8; ++j) {
    const int c = lane + 32*j;
    fr[c] = (__bf16)((v[j] - mean) * rstd * w[c] + bb[c]);
  }
}

// ---------------- avg-pool (ps x ps) + LayerNorm -> f tokens --------------
__global__ __launch_bounds__(32)
void pool_ln_kernel(const float* __restrict__ xp, const float* __restrict__ rw,
                    const float* __restrict__ rb, __bf16* __restrict__ f,
                    int ps, int nside, int tok_off) {
  const int lane = threadIdx.x;
  const int t = blockIdx.x, b = blockIdx.y;
  const int ti = t / nside, tj = t % nside;
  const float* xb = xp + (size_t)b * CH * HWSZ;
  const float inv = 1.f / (float)(ps * ps);
  float v[8], sum = 0.f;
  #pragma unroll
  for (int j = 0; j < 8; ++j) {
    const int c = lane + 32*j;
    const float* base = xb + (size_t)c * HWSZ + (ti*ps) * WW + tj*ps;
    float s = 0.f;
    for (int dy = 0; dy < ps; ++dy)
      for (int dx = 0; dx < ps; ++dx) s += base[dy*WW + dx];
    v[j] = s * inv; sum += v[j];
  }
  const float mean = wave_sum32(sum) * (1.f/CH);
  float vs = 0.f;
  #pragma unroll
  for (int j = 0; j < 8; ++j) { float d = v[j] - mean; vs += d*d; }
  const float rstd = rsqrtf(wave_sum32(vs) * (1.f/CH) + 1e-5f);
  __bf16* fr = f + ((size_t)b * SPAD + tok_off + t) * CH;
  #pragma unroll
  for (int j = 0; j < 8; ++j) {
    const int c = lane + 32*j;
    fr[c] = (__bf16)((v[j] - mean) * rstd * rw[c] + rb[c]);
  }
}

// ---------------- QKV:  qkv[b][s][j] = f @ in_proj_w^T + b (bf16 out) ------
__global__ __launch_bounds__(32)
void qkv_gemm_kernel(const __bf16* __restrict__ f, const float* __restrict__ w,
                     const float* __restrict__ bias, __bf16* __restrict__ qkv) {
  const int n0 = blockIdx.x * 64, m0 = blockIdx.y * 16, b = blockIdx.z;
  v8f acc[4] = {};
  gemm_tiles_bf16_wf32<4>(f + (size_t)b * SPAD * CH, CH, w, CH, m0, n0, CH, acc);
  const int lane = threadIdx.x, half = lane >> 4, idx = lane & 15;
  __bf16* out = qkv + (size_t)b * SPAD * (3*CH);
  #pragma unroll
  for (int t = 0; t < 4; ++t) {
    const float bi = bias[n0 + t*16 + idx];
    #pragma unroll
    for (int r = 0; r < 8; ++r) {
      const int s = m0 + half*8 + r;
      if (s < SEQ) out[(size_t)s * (3*CH) + n0 + t*16 + idx] = (__bf16)(acc[t][r] + bi);
    }
  }
}

// ---------------- flash attention with block-sparse skip ------------------
__global__ __launch_bounds__(32)
void attn_kernel(const __bf16* __restrict__ qkv,
                 const unsigned char* __restrict__ mask,
                 __bf16* __restrict__ o) {
  const int lane = threadIdx.x, half = lane >> 4, idx = lane & 15;
  const int q0 = blockIdx.x * 16;     // only local queries (144 tiles)
  const int h  = blockIdx.y;
  const int b  = blockIdx.z;
  const __bf16* base = qkv + (size_t)b * SPAD * (3*CH);

#if HAVE_ASYNC_LDS
  __shared__ __bf16 kbuf[32 * DHEAD];   // 2 KB: token-major K tile
  __shared__ __bf16 vbuf[32 * DHEAD];   // 2 KB: token-major V tile
#endif

  v16bf qb;                           // Q as B-frag: col=query, k=g*8+i / 16+g*8+i (dh)
  {
    const __bf16* qp = base + (size_t)(q0 + idx) * (3*CH) + h*DHEAD + half*8;
    #pragma unroll
    for (int i = 0; i < 16; ++i) qb[i] = qp[(i >> 3)*16 + (i & 7)];
  }
  const unsigned char* mrow = mask + (size_t)(q0 + idx) * SEQ;

  float mrun = -1e30f, lrun = 0.f;
  v8f o0 = {}, o1 = {};
  const float scale = 0.17677669529663687f;   // 1/sqrt(32)

  for (int kv0 = 0; kv0 < SEQ; kv0 += 32) {
    // mask probe first: skip fully blocked 32-key blocks (90% sparsity)
    bool mk[16]; bool anyv = false;
    #pragma unroll
    for (int r = 0; r < 8; ++r) {
      const int kg0 = kv0 + half*8 + r;
      const int kg1 = kv0 + 16 + half*8 + r;
      mk[r]     = (kg0 < SEQ) && (mrow[kg0] == 0);
      mk[r + 8] = (kg1 < SEQ) && (mrow[kg1] == 0);
      anyv = anyv || mk[r] || mk[r + 8];
    }
    if (__ballot(anyv) == 0ull) continue;

    v16bf ka0, ka1, vb0, vb1;
#if HAVE_ASYNC_LDS
    {
      // lane L stages token kv0+L: 16 bf16 (dims 0-7, 16-23) of K and of V (2x b128 each)
      const __bf16* krow = base + (size_t)(kv0 + lane) * (3*CH) + CH + h*DHEAD;
      const __bf16* vrow = krow + CH;
      __builtin_amdgcn_global_load_async_to_lds_b128(
          (gv4i_p)(void*)krow,        (lv4i_p)(void*)&kbuf[lane*DHEAD],      0, 0);
      __builtin_amdgcn_global_load_async_to_lds_b128(
          (gv4i_p)(void*)(krow + 16), (lv4i_p)(void*)&kbuf[lane*DHEAD + 16], 0, 0);
      __builtin_amdgcn_global_load_async_to_lds_b128(
          (gv4i_p)(void*)vrow,        (lv4i_p)(void*)&vbuf[lane*DHEAD],      0, 0);
      __builtin_amdgcn_global_load_async_to_lds_b128(
          (gv4i_p)(void*)(vrow + 16), (lv4i_p)(void*)&vbuf[lane*DHEAD + 16], 0, 0);
      __builtin_amdgcn_s_wait_asynccnt(0);
    }
    #pragma unroll
    for (int i = 0; i < 8; ++i) {       // K A-frags from LDS
      ka0[i]     = kbuf[idx*DHEAD + half*8 + i];
      ka0[i + 8] = kbuf[idx*DHEAD + 16 + half*8 + i];
      ka1[i]     = kbuf[(16 + idx)*DHEAD + half*8 + i];
      ka1[i + 8] = kbuf[(16 + idx)*DHEAD + 16 + half*8 + i];
    }
    #pragma unroll
    for (int i = 0; i < 16; ++i) {      // V B-frags from LDS
      vb0[i] = vbuf[((i >> 3)*16 + half*8 + (i & 7))*DHEAD + idx];
      vb1[i] = vbuf[((i >> 3)*16 + half*8 + (i & 7))*DHEAD + 16 + idx];
    }
#else
    {
      const __bf16* kp0 = base + (size_t)(kv0 + idx) * (3*CH) + CH + h*DHEAD;
      const __bf16* kp1 = kp0 + (size_t)16 * (3*CH);
      #pragma unroll
      for (int i = 0; i < 8; ++i) {
        ka0[i]     = kp0[half*8 + i];
        ka0[i + 8] = kp0[16 + half*8 + i];
        ka1[i]     = kp1[half*8 + i];
        ka1[i + 8] = kp1[16 + half*8 + i];
      }
      const __bf16* vp = base + (size_t)(kv0 + half*8) * (3*CH) + 2*CH + h*DHEAD;
      #pragma unroll
      for (int i = 0; i < 16; ++i) {
        vb0[i] = vp[(size_t)((i >> 3)*16 + (i & 7)) * (3*CH) + idx];
        vb1[i] = vp[(size_t)((i >> 3)*16 + (i & 7)) * (3*CH) + 16 + idx];
      }
    }
#endif

    v8f s0 = {}, s1 = {};
    s0 = __builtin_amdgcn_wmma_f32_16x16x32_bf16(false, ka0, false, qb, (short)0, s0, false, false);
    s1 = __builtin_amdgcn_wmma_f32_16x16x32_bf16(false, ka1, false, qb, (short)0, s1, false, false);

    // per-lane: one query column, 16 key scores (rows g*8+r and 16+g*8+r)
    float sc[16];
    #pragma unroll
    for (int r = 0; r < 8; ++r) {
      sc[r]     = mk[r]     ? s0[r] * scale : -1e30f;
      sc[r + 8] = mk[r + 8] ? s1[r] * scale : -1e30f;
    }
    float bm = sc[0];
    #pragma unroll
    for (int i = 1; i < 16; ++i) bm = fmaxf(bm, sc[i]);
    bm = fmaxf(bm, __shfl_xor(bm, 16));            // combine lane pair (same q)
    const float mnew  = fmaxf(mrun, bm);
    const float alpha = __expf(mrun - mnew);
    float ps[16], psum = 0.f;
    #pragma unroll
    for (int i = 0; i < 16; ++i) { ps[i] = __expf(sc[i] - mnew); psum += ps[i]; }
    psum += __shfl_xor(psum, 16);
    lrun = lrun * alpha + psum;
    mrun = mnew;

    // rescale O accumulators: row q = g*8+r, factor lives in lane (g*8+r)
    #pragma unroll
    for (int r = 0; r < 8; ++r) {
      const float af = __shfl(alpha, half*8 + r);
      o0[r] *= af; o1[r] *= af;
    }

    // P already sits in A-frag order: i=r -> k=g*8+r, i=8+r -> k=16+g*8+r
    v16bf pa;
    #pragma unroll
    for (int i = 0; i < 16; ++i) pa[i] = (__bf16)ps[i];

    o0 = __builtin_amdgcn_wmma_f32_16x16x32_bf16(false, pa, false, vb0, (short)0, o0, false, false);
    o1 = __builtin_amdgcn_wmma_f32_16x16x32_bf16(false, pa, false, vb1, (short)0, o1, false, false);
  }

  __bf16* ob = o + ((size_t)b * LOCAL + q0) * CH + h*DHEAD;
  #pragma unroll
  for (int r = 0; r < 8; ++r) {
    const float rl  = __shfl(lrun, half*8 + r);
    const float inv = 1.0f / rl;
    const int   t   = half*8 + r;
    ob[(size_t)t * CH + idx]      = (__bf16)(o0[r] * inv);
    ob[(size_t)t * CH + 16 + idx] = (__bf16)(o1[r] * inv);
  }
}

// ---------------- out-proj (local tokens only), f32 out -------------------
__global__ __launch_bounds__(32)
void outproj_kernel(const __bf16* __restrict__ o, const float* __restrict__ w,
                    const float* __restrict__ bias, float* __restrict__ lo) {
  const int n0 = blockIdx.x * 64, m0 = blockIdx.y * 16, b = blockIdx.z;
  v8f acc[4] = {};
  gemm_tiles_bf16_wf32<4>(o + (size_t)b * LOCAL * CH, CH, w, CH, m0, n0, CH, acc);
  const int lane = threadIdx.x, half = lane >> 4, idx = lane & 15;
  float* lob = lo + (size_t)b * LOCAL * CH;
  #pragma unroll
  for (int t = 0; t < 4; ++t) {
    const float bi = bias[n0 + t*16 + idx];
    #pragma unroll
    for (int r = 0; r < 8; ++r)
      lob[(size_t)(m0 + half*8 + r) * CH + n0 + t*16 + idx] = acc[t][r] + bi;
  }
}

// ---------------- LayerNorm over contiguous rows (lo -> hln bf16) ---------
__global__ __launch_bounds__(32)
void ln_rows_kernel(const float* __restrict__ in, const float* __restrict__ w,
                    const float* __restrict__ bb, __bf16* __restrict__ out) {
  const int lane = threadIdx.x;
  const int t = blockIdx.x, b = blockIdx.y;
  const float* row = in + ((size_t)b * LOCAL + t) * CH;
  float v[8], sum = 0.f;
  #pragma unroll
  for (int j = 0; j < 8; ++j) { v[j] = row[lane + 32*j]; sum += v[j]; }
  const float mean = wave_sum32(sum) * (1.f/CH);
  float vs = 0.f;
  #pragma unroll
  for (int j = 0; j < 8; ++j) { float d = v[j] - mean; vs += d*d; }
  const float rstd = rsqrtf(wave_sum32(vs) * (1.f/CH) + 1e-5f);
  __bf16* orow = out + ((size_t)b * LOCAL + t) * CH;
  #pragma unroll
  for (int j = 0; j < 8; ++j) {
    const int c = lane + 32*j;
    orow[c] = (__bf16)((v[j] - mean) * rstd * w[c] + bb[c]);
  }
}

// ---------------- MLP fc1 + exact GELU -> mid (bf16) ----------------------
__global__ __launch_bounds__(32)
void mlp1_kernel(const __bf16* __restrict__ hln, const float* __restrict__ w1,
                 const float* __restrict__ b1, __bf16* __restrict__ mid) {
  const int m0 = blockIdx.y * 16, b = blockIdx.z;   // N = 64 -> single wave in N
  v8f acc[4] = {};
  gemm_tiles_bf16_wf32<4>(hln + (size_t)b * LOCAL * CH, CH, w1, CH, m0, 0, CH, acc);
  const int lane = threadIdx.x, half = lane >> 4, idx = lane & 15;
  __bf16* mb = mid + (size_t)b * LOCAL * BOT;
  #pragma unroll
  for (int t = 0; t < 4; ++t) {
    const float bi = b1[t*16 + idx];
    #pragma unroll
    for (int r = 0; r < 8; ++r) {
      const float z = acc[t][r] + bi;
      const float g = 0.5f * z * (1.0f + erff(z * 0.70710678118654752f));
      mb[(size_t)(m0 + half*8 + r) * BOT + t*16 + idx] = (__bf16)g;
    }
  }
}

// ---------------- MLP fc2 + lo + x, transposed store to d_out -------------
__global__ __launch_bounds__(32)
void mlp2_final_kernel(const __bf16* __restrict__ mid, const float* __restrict__ w2,
                       const float* __restrict__ b2, const float* __restrict__ lo,
                       const float* __restrict__ x, float* __restrict__ out) {
  const int n0 = blockIdx.x * 64, m0 = blockIdx.y * 16, b = blockIdx.z;
  v8f acc[4] = {};
  gemm_tiles_bf16_wf32<4>(mid + (size_t)b * LOCAL * BOT, BOT, w2, BOT, m0, n0, BOT, acc);
  const int lane = threadIdx.x, half = lane >> 4, idx = lane & 15;
  const float* lob = lo + (size_t)b * LOCAL * CH;
  #pragma unroll
  for (int t = 0; t < 4; ++t) {
    const int c = n0 + t*16 + idx;
    const float bi = b2[c];
    const float* xc = x   + ((size_t)b * CH + c) * HWSZ;
    float*       oc = out + ((size_t)b * CH + c) * HWSZ;
    #pragma unroll
    for (int r = 0; r < 8; ++r) {
      const int tk = m0 + half*8 + r;
      oc[tk] = acc[t][r] + bi + lob[(size_t)tk * CH + c] + xc[tk];
    }
  }
}

// ---------------------------------------------------------------------------
extern "C" void kernel_launch(void* const* d_in, const int* in_sizes, int n_in,
                              void* d_out, int out_size, void* d_ws, size_t ws_size,
                              hipStream_t stream) {
  (void)in_sizes; (void)n_in; (void)out_size; (void)ws_size;
  const float* x          = (const float*)d_in[0];
  const float* conv_w     = (const float*)d_in[1];
  const float* conv_b     = (const float*)d_in[2];
  const float* ln_local_w = (const float*)d_in[3];
  const float* ln_local_b = (const float*)d_in[4];
  const float* ln_reg_w   = (const float*)d_in[5];
  const float* ln_reg_b   = (const float*)d_in[6];
  const float* in_proj_w  = (const float*)d_in[7];
  const float* in_proj_b  = (const float*)d_in[8];
  const float* out_proj_w = (const float*)d_in[9];
  const float* out_proj_b = (const float*)d_in[10];
  const float* ln_out_w   = (const float*)d_in[11];
  const float* ln_out_b   = (const float*)d_in[12];
  const float* mlp_w1     = (const float*)d_in[13];
  const float* mlp_b1     = (const float*)d_in[14];
  const float* mlp_w2     = (const float*)d_in[15];
  const float* mlp_b2     = (const float*)d_in[16];
  const unsigned char* attn_mask = (const unsigned char*)d_in[17];
  float* out = (float*)d_out;

  // workspace carve-up (~50 MB)
  char* w = (char*)d_ws;
  float*  xp  = (float*)w;  w += (size_t)BATCH*CH*HWSZ*sizeof(float);
  __bf16* f   = (__bf16*)w; w += (size_t)BATCH*SPAD*CH*sizeof(__bf16);
  __bf16* qkv = (__bf16*)w; w += (size_t)BATCH*SPAD*3*CH*sizeof(__bf16);
  __bf16* oat = (__bf16*)w; w += (size_t)BATCH*LOCAL*CH*sizeof(__bf16);
  float*  lo  = (float*)w;  w += (size_t)BATCH*LOCAL*CH*sizeof(float);
  __bf16* hln = (__bf16*)w; w += (size_t)BATCH*LOCAL*CH*sizeof(__bf16);
  __bf16* mid = (__bf16*)w; w += (size_t)BATCH*LOCAL*BOT*sizeof(__bf16);

  dim3 blk(32);
  // 1) 1x1 conv as GEMM (M=256 out-ch, N=2304 pix, K=256), 64x16 per wave
  conv1x1_kernel<<<dim3(HWSZ/16, CH/64, BATCH), blk, 0, stream>>>(x, conv_w, conv_b, xp);
  // 2) token features: local LN + pooled(4) + pooled(8)
  ln_local_kernel<<<dim3(HWSZ, BATCH), blk, 0, stream>>>(xp, ln_local_w, ln_local_b, f);
  pool_ln_kernel<<<dim3(144, BATCH), blk, 0, stream>>>(xp, ln_reg_w,      ln_reg_b,      f, 4, 12, HWSZ);
  pool_ln_kernel<<<dim3(36,  BATCH), blk, 0, stream>>>(xp, ln_reg_w + CH, ln_reg_b + CH, f, 8, 6,  HWSZ + 144);
  // 3) QKV projection (16x64 per wave)
  qkv_gemm_kernel<<<dim3(3*CH/64, SPAD/16, BATCH), blk, 0, stream>>>(f, in_proj_w, in_proj_b, qkv);
  // 4) block-sparse flash attention (only local queries needed downstream)
  attn_kernel<<<dim3(LOCAL/16, NHEAD, BATCH), blk, 0, stream>>>(qkv, attn_mask, oat);
  // 5) out projection on local tokens (16x64 per wave)
  outproj_kernel<<<dim3(CH/64, LOCAL/16, BATCH), blk, 0, stream>>>(oat, out_proj_w, out_proj_b, lo);
  // 6) LN -> MLP -> residual -> transpose(+x) into d_out
  ln_rows_kernel<<<dim3(LOCAL, BATCH), blk, 0, stream>>>(lo, ln_out_w, ln_out_b, hln);
  mlp1_kernel<<<dim3(1, LOCAL/16, BATCH), blk, 0, stream>>>(hln, mlp_w1, mlp_b1, mid);
  mlp2_final_kernel<<<dim3(CH/64, LOCAL/16, BATCH), blk, 0, stream>>>(mid, mlp_w2, mlp_b2, lo, x, out);
}
